// NonLocalRGBIR_69810398429277
// MI455X (gfx1250) — matrix operand order repaired and checked by reading, not compiled, not run on hardware
//
#include <hip/hip_runtime.h>

constexpr int kB  = 8,  kC  = 64, kIC = 32;
constexpr int kH  = 80, kW  = 80, kN  = 6400;
constexpr int kHp = 40, kWp = 40, kM  = 1600;
constexpr int kGC = 6400, kGH = 3200;
constexpr float kEps = 1e-5f;

typedef __attribute__((ext_vector_type(16))) __bf16 v16bf;
typedef __attribute__((ext_vector_type(8)))  __bf16 v8bf;
typedef __attribute__((ext_vector_type(8)))  float  v8f;

// --------------------------- CDNA5 async helpers ---------------------------
// GVS mode: mem = SADDR(64) + VADDR(32-bit byte offset) + INST_OFFSET,
// LDS dst = LDS_BASE + VDST(32-bit byte offset) + INST_OFFSET.
// INST_OFFSET shifts BOTH addresses (ISA 15.18.3), handy for paired b128s.
#define ASYNC_B128(ldsoff, goff, sbase, imm)                                   \
  asm volatile("global_load_async_to_lds_b128 %0, %1, %2 offset:%3"            \
               :: "v"(ldsoff), "v"(goff), "s"(sbase), "n"(imm) : "memory")
#define ASYNC_B64(ldsoff, goff, sbase, imm)                                    \
  asm volatile("global_load_async_to_lds_b64 %0, %1, %2 offset:%3"             \
               :: "v"(ldsoff), "v"(goff), "s"(sbase), "n"(imm) : "memory")
#define WAIT_ASYNC0()                                                          \
  asm volatile("s_wait_asynccnt 0x0" ::: "memory")

// ---------------------------------------------------------------------------
// Geometry hidden layer: h_t[m][gh] = relu(w1[gh]·pos[:,m] + b1[gh]), bf16.
// ---------------------------------------------------------------------------
__global__ void __launch_bounds__(256)
geom_h_kernel(const float* __restrict__ pos, const float* __restrict__ w1,
              const float* __restrict__ b1, __bf16* __restrict__ ht) {
  int idx = blockIdx.x * 256 + threadIdx.x;
  if (idx >= kM * kGH) return;
  int m = idx / kGH, gh = idx % kGH;
  float p0 = pos[m], p1 = pos[kM + m];
  float h = fmaf(w1[gh * 2 + 0], p0, fmaf(w1[gh * 2 + 1], p1, b1[gh]));
  ht[(size_t)m * kGH + gh] = (__bf16)fmaxf(h, 0.0f);
}

// w2 fp32 -> bf16 (so the bias GEMM A operand is async-to-LDS copyable).
__global__ void __launch_bounds__(256)
w2cvt_kernel(const float* __restrict__ w2, __bf16* __restrict__ w2bf) {
  int idx = blockIdx.x * 256 + threadIdx.x;
  if (idx < kGC * kGH) w2bf[idx] = (__bf16)w2[idx];
}

// ---------------------------------------------------------------------------
// Geometry bias GEMM: bias[gc][m] = w2[gc][:]·h[:][m] + b2[gc]
// Block tile 128(gc) x 64(m); K steps of 32; A/B tiles double-buffered in LDS
// via global_load_async_to_lds_b128; 8 waves each compute a 32x32 sub-tile.
// LDS: Abuf[2][128][32]bf16 (16KB) + Bbuf[2][64][32]bf16 (8KB) = 24KB dynamic.
// ---------------------------------------------------------------------------
__global__ void __launch_bounds__(256)
bias_gemm_kernel(const __bf16* __restrict__ w2bf, const float* __restrict__ b2,
                 const __bf16* __restrict__ ht, float* __restrict__ bias) {
  extern __shared__ __align__(128) char smem[];
  const unsigned ABUF = 0, BBUF = 16384;      // Abuf[2]@0 (8KB each), Bbuf[2]@16K (4KB each)
  const int t    = threadIdx.x;
  const int lane = t & 31, wave = t >> 5;
  const int half = lane >> 4, lm = lane & 15;
  const int gcb  = (blockIdx.x / 25) * 128;
  const int mb   = (blockIdx.x % 25) * 64;
  const int wgc  = (wave >> 1) * 32;          // 0,32,64,96
  const int wm   = (wave & 1) * 32;           // 0,32

  // per-thread async-copy assignments
  const unsigned arow = t >> 1, akp = (t & 1);            // A: 32B/thread
  const unsigned avoff0 = (unsigned)(gcb + arow) * (kGH * 2) + akp * 32;
  const unsigned aldso  = (unsigned)t * 32;
  const unsigned brow = t >> 2, bkp = (t & 3);            // B: 16B/thread
  const unsigned bvoff0 = (unsigned)(mb + brow) * (kGH * 2) + bkp * 16;
  const unsigned bldso  = (unsigned)t * 16;

  v8f D[2][2];
#pragma unroll
  for (int sA = 0; sA < 2; ++sA) {
#pragma unroll
    for (int r = 0; r < 8; ++r) {
      float bv = b2[gcb + wgc + 16 * sA + r + 8 * half];
      D[sA][0][r] = bv; D[sA][1][r] = bv;
    }
  }

  // prologue: stage k-tile 0 into buffer 0
  ASYNC_B128(ABUF + aldso, avoff0, w2bf, 0);
  ASYNC_B128(ABUF + aldso, avoff0, w2bf, 16);
  ASYNC_B128(BBUF + bldso, bvoff0, ht, 0);

  const int NK = kGH / 32;                    // 100
  for (int kt = 0; kt < NK; ++kt) {
    WAIT_ASYNC0();
    __syncthreads();                          // tile kt resident for all waves
    const unsigned ab = ABUF + (kt & 1) * 8192;
    const unsigned bb = BBUF + (kt & 1) * 4096;
    v16bf Af[2], Bf[2];
#pragma unroll
    for (int s = 0; s < 2; ++s) {
      unsigned row = wgc + 16 * s + lm;
      const __bf16* ap = (const __bf16*)(smem + ab + row * 64 + 32 * half);
      v8bf a0 = *(const v8bf*)ap;
      v8bf a1 = *(const v8bf*)(ap + 8);
#pragma unroll
      for (int i = 0; i < 8; ++i) { Af[s][i] = a0[i]; Af[s][i + 8] = a1[i]; }
      unsigned mrow = wm + 16 * s + lm;
      Bf[s] = *(const v16bf*)(smem + bb + mrow * 64 + 32 * half);
    }
    __syncthreads();                          // all reads of buffer kt done
    if (kt + 1 < NK) {                        // overlap next tile DMA w/ math
      unsigned ko = (unsigned)(kt + 1) * 64;
      unsigned dst = ABUF + ((kt + 1) & 1) * 8192 + aldso;
      ASYNC_B128(dst, avoff0 + ko, w2bf, 0);
      ASYNC_B128(dst, avoff0 + ko, w2bf, 16);
      ASYNC_B128(BBUF + ((kt + 1) & 1) * 4096 + bldso, bvoff0 + ko, ht, 0);
    }
#pragma unroll
    for (int sA = 0; sA < 2; ++sA)
#pragma unroll
      for (int sB = 0; sB < 2; ++sB)
        D[sA][sB] = __builtin_amdgcn_wmma_f32_16x16x32_bf16(
            false, Af[sA], false, Bf[sB], (short)0, D[sA][sB], false, false);
  }

#pragma unroll
  for (int sA = 0; sA < 2; ++sA)
#pragma unroll
    for (int sB = 0; sB < 2; ++sB)
#pragma unroll
      for (int r = 0; r < 8; ++r)
        bias[(size_t)(gcb + wgc + 16 * sA + r + 8 * half) * kM
             + mb + wm + 16 * sB + lm] = D[sA][sB][r];
}

// ---------------------------------------------------------------------------
// theta[n][ic] (bf16): 1x1 conv over C=64 channels.
// ---------------------------------------------------------------------------
__global__ void __launch_bounds__(256)
theta_kernel(const float* __restrict__ x, const float* __restrict__ tw,
             const float* __restrict__ tb, __bf16* __restrict__ theta) {
  int idx = blockIdx.x * 256 + threadIdx.x;
  if (idx >= kB * kN) return;
  int b = idx / kN, n = idx % kN;
  float xv[kC];
#pragma unroll
  for (int c = 0; c < kC; ++c) xv[c] = x[((size_t)b * kC + c) * kN + n];
  for (int o = 0; o < kIC; ++o) {
    float acc = tb[o];
#pragma unroll
    for (int c = 0; c < kC; ++c) acc = fmaf(tw[o * kC + c], xv[c], acc);
    theta[((size_t)b * kN + n) * kIC + o] = (__bf16)acc;
  }
}

// ---------------------------------------------------------------------------
// Fused maxpool2 + phi[key][ic] + g_t[ic][key] (both bf16).
// ---------------------------------------------------------------------------
__global__ void __launch_bounds__(256)
phig_kernel(const float* __restrict__ x, const float* __restrict__ pw,
            const float* __restrict__ pb, const float* __restrict__ gw,
            const float* __restrict__ gb, __bf16* __restrict__ phi,
            __bf16* __restrict__ gt) {
  int idx = blockIdx.x * 256 + threadIdx.x;
  if (idx >= kB * kM) return;
  int b = idx / kM, m = idx % kM;
  int py = m / kWp, px = m % kWp;
  float pool[kC];
#pragma unroll
  for (int c = 0; c < kC; ++c) {
    size_t base = (((size_t)b * kC + c) * kH + 2 * py) * kW + 2 * px;
    float v0 = x[base],      v1 = x[base + 1];
    float v2 = x[base + kW], v3 = x[base + kW + 1];
    pool[c] = fmaxf(fmaxf(v0, v1), fmaxf(v2, v3));
  }
  for (int o = 0; o < kIC; ++o) {
    float ap = pb[o], ag = gb[o];
#pragma unroll
    for (int c = 0; c < kC; ++c) {
      ap = fmaf(pw[o * kC + c], pool[c], ap);
      ag = fmaf(gw[o * kC + c], pool[c], ag);
    }
    phi[((size_t)b * kM + m) * kIC + o] = (__bf16)ap;
    gt[((size_t)b * kIC + o) * kM + m]  = (__bf16)ag;
  }
}

// ---------------------------------------------------------------------------
// Flash attention. All 8 waves of a block share batch index b, so phi/g key
// chunks (32 keys) are staged once per block in double-buffered LDS via async
// DMA. Per wave: S = theta x phi + bias (2 WMMA, bias as C), online softmax,
// P through per-wave LDS bounce, O += P x g^T (2 WMMA).
// Dynamic LDS: pbuf[2]:4KB @0, gbuf[2]:4KB @4096, plds 8KB @8192 = 16KB.
// ---------------------------------------------------------------------------
__global__ void __launch_bounds__(256)
attn_kernel(const __bf16* __restrict__ theta, const __bf16* __restrict__ phi,
            const __bf16* __restrict__ gt, const float* __restrict__ bias,
            float* __restrict__ attn_out) {
  extern __shared__ __align__(128) char smem[];
  const unsigned PBUF = 0, GBUF = 4096, PLDS = 8192;
  const int t    = threadIdx.x;
  const int wave = t >> 5, lane = t & 31;
  const int half = lane >> 4, lm = lane & 15;
  const int tile = blockIdx.x * 8 + wave;     // 0..3199 (exact)
  const int b    = tile / (kN / 16);          // uniform across block (50 blk/b)
  const int qb   = (tile % (kN / 16)) * 16;

  const __bf16* pbase = phi + (size_t)b * kM * kIC;
  const __bf16* gbase = gt + (size_t)b * kIC * kM;

  // per-thread async assignments (8B each; chunk = 2KB per operand)
  const unsigned pvoff0 = (unsigned)t * 8;                // contiguous [key][ic]
  const unsigned pldso  = PBUF + (unsigned)t * 8;
  const unsigned gic = t >> 3, gkp = t & 7;               // gbuf[ic][32 keys]
  const unsigned gvoff0 = gic * (kM * 2) + gkp * 8;
  const unsigned gldso  = GBUF + gic * 64 + gkp * 8;

  // A operand: theta rows qb..qb+15
  const __bf16* tp = theta + ((size_t)b * kN + qb + lm) * kIC + half * 16;
  v8bf t0 = *(const v8bf*)tp, t1 = *(const v8bf*)(tp + 8);
  v16bf Aq;
#pragma unroll
  for (int i = 0; i < 8; ++i) { Aq[i] = t0[i]; Aq[i + 8] = t1[i]; }

  v8f O0 = {}, O1 = {};
  float mrun[8], lrun[8];
#pragma unroll
  for (int r = 0; r < 8; ++r) { mrun[r] = -3.0e38f; lrun[r] = 0.0f; }

  // prologue: stage key-chunk 0
  ASYNC_B64(pldso, pvoff0, pbase, 0);
  ASYNC_B64(gldso, gvoff0, gbase, 0);

  const int NK = kM / 32;                     // 50
  for (int kt = 0; kt < NK; ++kt) {
    const int kb = kt * 32;
    WAIT_ASYNC0();
    __syncthreads();                          // chunk kt resident
    const unsigned pb_ = (kt & 1) * 2048;
    v16bf B0 = *(const v16bf*)(smem + PBUF + pb_ + (unsigned)lm * 64 + 32 * half);
    v16bf B1 = *(const v16bf*)(smem + PBUF + pb_ + (unsigned)(16 + lm) * 64 + 32 * half);
    v16bf Bg0 = *(const v16bf*)(smem + GBUF + pb_ + (unsigned)lm * 64 + 32 * half);
    v16bf Bg1 = *(const v16bf*)(smem + GBUF + pb_ + (unsigned)(16 + lm) * 64 + 32 * half);
    __syncthreads();                          // chunk kt reads done
    if (kt + 1 < NK) {                        // overlap next chunk DMA
      unsigned nb = ((kt + 1) & 1) * 2048;
      ASYNC_B64(pldso + nb, pvoff0 + (unsigned)(kt + 1) * 2048, pbase, 0);
      ASYNC_B64(gldso + nb, gvoff0 + (unsigned)(kt + 1) * 64, gbase, 0);
    }
    // bias tile as C operand
    v8f C0, C1;
#pragma unroll
    for (int r = 0; r < 8; ++r) {
      const float* bb = bias + (size_t)(qb + r + 8 * half) * kM + kb + lm;
      C0[r] = bb[0]; C1[r] = bb[16];
    }
    v8f S0 = __builtin_amdgcn_wmma_f32_16x16x32_bf16(false, Aq, false, B0,
                                                     (short)0, C0, false, false);
    v8f S1 = __builtin_amdgcn_wmma_f32_16x16x32_bf16(false, Aq, false, B1,
                                                     (short)0, C1, false, false);
    // online softmax
    float P0[8], P1[8];
#pragma unroll
    for (int r = 0; r < 8; ++r) {
      float v = fmaxf(S0[r], S1[r]);
      v = fmaxf(v, __shfl_xor(v, 1)); v = fmaxf(v, __shfl_xor(v, 2));
      v = fmaxf(v, __shfl_xor(v, 4)); v = fmaxf(v, __shfl_xor(v, 8));
      float mn = fmaxf(mrun[r], v);
      float sc = __expf(mrun[r] - mn);
      mrun[r] = mn;
      O0[r] *= sc; O1[r] *= sc; lrun[r] *= sc;
      float e0 = __expf(S0[r] - mn), e1 = __expf(S1[r] - mn);
      P0[r] = e0; P1[r] = e1;
      float s = e0 + e1;
      s += __shfl_xor(s, 1); s += __shfl_xor(s, 2);
      s += __shfl_xor(s, 4); s += __shfl_xor(s, 8);
      lrun[r] += s;
    }
    // C/D-layout -> A-layout via per-wave LDS bounce
    __syncthreads();
    {
      __bf16* pl = (__bf16*)(smem + PLDS + (unsigned)wave * 1024);
#pragma unroll
      for (int r = 0; r < 8; ++r) {
        pl[(r + 8 * half) * 32 + lm]      = (__bf16)P0[r];
        pl[(r + 8 * half) * 32 + 16 + lm] = (__bf16)P1[r];
      }
    }
    __syncthreads();
    const __bf16* lp =
        (const __bf16*)(smem + PLDS + (unsigned)wave * 1024) + lm * 32 + 16 * half;
    v8bf q0 = *(const v8bf*)lp, q1 = *(const v8bf*)(lp + 8);
    v16bf Ap;
#pragma unroll
    for (int i = 0; i < 8; ++i) { Ap[i] = q0[i]; Ap[i + 8] = q1[i]; }
    O0 = __builtin_amdgcn_wmma_f32_16x16x32_bf16(false, Ap, false, Bg0,
                                                 (short)0, O0, false, false);
    O1 = __builtin_amdgcn_wmma_f32_16x16x32_bf16(false, Ap, false, Bg1,
                                                 (short)0, O1, false, false);
  }
#pragma unroll
  for (int r = 0; r < 8; ++r) {
    float inv = 1.0f / lrun[r];
    int q = qb + r + 8 * half;
    attn_out[((size_t)b * kN + q) * kIC + lm]      = O0[r] * inv;
    attn_out[((size_t)b * kN + q) * kIC + 16 + lm] = O1[r] * inv;
  }
}

// ---------------------------------------------------------------------------
// BN stats: one block per channel; y = W·attn + b recomputed on the fly.
// ---------------------------------------------------------------------------
__global__ void __launch_bounds__(256)
stats_kernel(const float* __restrict__ attn, const float* __restrict__ Ww,
             const float* __restrict__ Wb, float* __restrict__ stats) {
  const int c = blockIdx.x;
  float wloc[kIC];
#pragma unroll
  for (int i = 0; i < kIC; ++i) wloc[i] = Ww[c * kIC + i];
  const float wb = Wb[c];
  float s = 0.0f, s2 = 0.0f;
  for (int t = threadIdx.x; t < kB * kN; t += 256) {
    const float* a = attn + (size_t)t * kIC;
    float y = wb;
#pragma unroll
    for (int i = 0; i < kIC; ++i) y = fmaf(wloc[i], a[i], y);
    s += y; s2 = fmaf(y, y, s2);
  }
  __shared__ float r1[256], r2[256];
  r1[threadIdx.x] = s; r2[threadIdx.x] = s2;
  __syncthreads();
  for (int st = 128; st > 0; st >>= 1) {
    if (threadIdx.x < st) {
      r1[threadIdx.x] += r1[threadIdx.x + st];
      r2[threadIdx.x] += r2[threadIdx.x + st];
    }
    __syncthreads();
  }
  if (threadIdx.x == 0) {
    const float cnt = (float)(kB * kN);
    float mean = r1[0] / cnt;
    float var  = r2[0] / cnt - mean * mean;
    stats[c] = mean;
    stats[kC + c] = 1.0f / sqrtf(var + kEps);
  }
}

// ---------------------------------------------------------------------------
// Finalize: recompute y, apply BN + residual -> out.
// ---------------------------------------------------------------------------
__global__ void __launch_bounds__(256)
final_kernel(const float* __restrict__ attn, const float* __restrict__ x,
             const float* __restrict__ Ww, const float* __restrict__ Wb,
             const float* __restrict__ bng, const float* __restrict__ bnb,
             const float* __restrict__ stats, float* __restrict__ out) {
  int idx = blockIdx.x * 256 + threadIdx.x;
  if (idx >= kB * kN) return;
  int b = idx / kN, n = idx % kN;
  float a[kIC];
#pragma unroll
  for (int i = 0; i < kIC; ++i) a[i] = attn[((size_t)b * kN + n) * kIC + i];
  for (int c = 0; c < kC; ++c) {
    float y = Wb[c];
#pragma unroll
    for (int i = 0; i < kIC; ++i) y = fmaf(Ww[c * kIC + i], a[i], y);
    float v = fmaf(bng[c] * stats[kC + c], y - stats[c], bnb[c])
            + x[((size_t)b * kC + c) * kN + n];
    out[((size_t)b * kC + c) * kN + n] = v;
  }
}

// ---------------------------------------------------------------------------
extern "C" void kernel_launch(void* const* d_in, const int* in_sizes, int n_in,
                              void* d_out, int out_size, void* d_ws, size_t ws_size,
                              hipStream_t stream) {
  (void)in_sizes; (void)n_in; (void)out_size; (void)ws_size;
  const float* xin[2] = { (const float*)d_in[0], (const float*)d_in[1] };
  const float* pos   = (const float*)d_in[2];
  const float* gp_w1 = (const float*)d_in[3];
  const float* gp_b1 = (const float*)d_in[4];
  const float* gp_w2 = (const float*)d_in[5];
  const float* gp_b2 = (const float*)d_in[6];

  auto al = [](size_t v) { return (v + 255) & ~(size_t)255; };
  char* base = (char*)d_ws;
  size_t off = 0;
  float*  bias = (float*)(base + off);  off += al((size_t)kN * kM * 4);
  __bf16* ht   = (__bf16*)(base + off); off += al((size_t)kM * kGH * 2);
  __bf16* w2bf = (__bf16*)(base + off); off += al((size_t)kGC * kGH * 2);
  __bf16* theta[2]; __bf16* phi[2]; __bf16* gt[2]; float* attn[2]; float* stats[2];
  for (int p = 0; p < 2; ++p) {
    theta[p] = (__bf16*)(base + off); off += al((size_t)kB * kN * kIC * 2);
    phi[p]   = (__bf16*)(base + off); off += al((size_t)kB * kM * kIC * 2);
    gt[p]    = (__bf16*)(base + off); off += al((size_t)kB * kIC * kM * 2);
    attn[p]  = (float*)(base + off);  off += al((size_t)kB * kN * kIC * 4);
    stats[p] = (float*)(base + off);  off += al(2 * kC * 4);
  }

  geom_h_kernel<<<(kM * kGH) / 256, 256, 0, stream>>>(pos, gp_w1, gp_b1, ht);
  w2cvt_kernel<<<(kGC * kGH) / 256, 256, 0, stream>>>(gp_w2, w2bf);
  bias_gemm_kernel<<<(kGC / 128) * (kM / 64), 256, 24576, stream>>>(w2bf, gp_b2,
                                                                    ht, bias);
  for (int p = 0; p < 2; ++p) {
    const int o = 7 + p * 12;
    const float* tw  = (const float*)d_in[o + 0];
    const float* tb  = (const float*)d_in[o + 1];
    const float* pw  = (const float*)d_in[o + 2];
    const float* pb  = (const float*)d_in[o + 3];
    const float* gw  = (const float*)d_in[o + 4];
    const float* gb  = (const float*)d_in[o + 5];
    const float* Ww  = (const float*)d_in[o + 6];
    const float* Wb  = (const float*)d_in[o + 7];
    const float* bng = (const float*)d_in[o + 8];
    const float* bnb = (const float*)d_in[o + 9];
    float* outp = (float*)d_out + (size_t)p * kB * kC * kN;

    theta_kernel<<<(kB * kN) / 256, 256, 0, stream>>>(xin[p], tw, tb, theta[p]);
    phig_kernel<<<(kB * kM) / 256, 256, 0, stream>>>(xin[p], pw, pb, gw, gb,
                                                     phi[p], gt[p]);
    attn_kernel<<<(kB * (kN / 16)) / 8, 256, 16384, stream>>>(theta[p], phi[p],
                                                              gt[p], bias, attn[p]);
    stats_kernel<<<kC, 256, 0, stream>>>(attn[p], Ww, Wb, stats[p]);
    final_kernel<<<(kB * kN) / 256, 256, 0, stream>>>(attn[p], xin[p], Ww, Wb,
                                                      bng, bnb, stats[p], outp);
  }
}
